// GlobalNode_4870492914030
// MI455X (gfx1250) — compile-verified, matrix operand order checked by reading, not run on hardware
//
#include <hip/hip_runtime.h>
#include <hip/hip_bf16.h>

typedef __bf16        v16bf __attribute__((ext_vector_type(16)));
typedef float         v8f   __attribute__((ext_vector_type(8)));
typedef unsigned int  v8u   __attribute__((ext_vector_type(8)));

#define EMB   256
#define WSTR  132   // W_feat LDS row stride in dwords: bank=(4*row+c)%64 conflict-free AND 16B-aligned runs
#define XSTR  132   // x-chunk LDS row stride in dwords (same property)

static __device__ __forceinline__ unsigned pack_bf16(float a, float b) {
    unsigned ua = __float_as_uint(a);
    ua += 0x7FFFu + ((ua >> 16) & 1u);          // round-to-nearest-even
    unsigned ub = __float_as_uint(b);
    ub += 0x7FFFu + ((ub >> 16) & 1u);
    return (ua >> 16) | (ub & 0xFFFF0000u);
}

static __device__ __forceinline__ float wred_sum(float v) {
#pragma unroll
    for (int off = 16; off > 0; off >>= 1) v += __shfl_xor(v, off, 32);
    return v;
}

static __device__ __forceinline__ float lrelu(float x) {
    return x > 0.0f ? x : 0.01f * x;
}

// LDS layout (dynamic): wt[256*WSTR] u32 | xch[16*XSTR] u32 | wmask[256] f32 |
// bfl[256] f32 | wbuf[16] f32 | red[8] f32 | scal[2] f32 | hbuf[512] f32
#define SMEM_BYTES (((EMB * WSTR) + (16 * XSTR)) * 4 + (256 + 256 + 16 + 8 + 2 + 512) * 4)

__global__ __launch_bounds__(256)
void global_attn_pool_kernel(const float* __restrict__ xg_prev,
                             const float* __restrict__ x,
                             const int*   __restrict__ bind,
                             const float* __restrict__ Wmask,
                             const float* __restrict__ Wfeat,
                             const float* __restrict__ bfeat,
                             const float* __restrict__ Wt,
                             const float* __restrict__ bt,
                             float*       __restrict__ out,
                             float*       __restrict__ gate_ws,
                             int N)
{
    extern __shared__ char smem[];
    unsigned* wt    = (unsigned*)smem;               // 256 x WSTR dwords (bf16 pairs, transposed W_feat: [n][k])
    unsigned* xch   = wt + EMB * WSTR;               // 16 x XSTR dwords (bf16 pairs of x chunk)
    float*    wmask = (float*)(xch + 16 * XSTR);     // 256
    float*    bfl   = wmask + 256;                   // 256
    float*    wbuf  = bfl + 256;                     // 16 per-row softmax weights
    float*    red   = wbuf + 16;                     // 8 wave partials
    float*    scal  = red + 8;                       // [0]=seg_max, [1]=inv_denom
    float*    hbuf  = scal + 2;                      // 512: [xg | xg_prev]

    const int tid  = threadIdx.x;
    const int lane = tid & 31;
    const int wid  = tid >> 5;
    const int b    = blockIdx.x;

    // ---- segment bounds via binary search on sorted batch_ind ----
    int s0, s1;
    {
        int lo = 0, hi = N;
        while (lo < hi) { int m = (lo + hi) >> 1; if (bind[m] < b) lo = m + 1; else hi = m; }
        s0 = lo; hi = N;
        while (lo < hi) { int m = (lo + hi) >> 1; if (bind[m] < b + 1) lo = m + 1; else hi = m; }
        s1 = lo;
    }

    // ---- stage weights in LDS ----
    wmask[tid] = Wmask[tid];
    bfl[tid]   = bfeat[tid];
    // transpose W_feat [k][n] -> wt[n][k] as packed bf16 pairs; thread owns column n = tid
    for (int k2 = 0; k2 < EMB; k2 += 2) {
        float a = Wfeat[k2 * EMB + tid];
        float c = Wfeat[(k2 + 1) * EMB + tid];
        wt[tid * WSTR + (k2 >> 1)] = pack_bf16(a, c);
    }
    __syncthreads();

    const int tA   = wid, tB = wid + 8;  // the two N-tiles this wave owns
    const int mrow = lane & 15;          // M for A-frag / N for B-frag / N for C
    const int hh   = lane >> 4;          // lane half

    // ---- preload loop-invariant B fragments (W_feat) into registers: 2 tiles x 8 k-steps ----
    v8u BfA[8], BfB[8];
#pragma unroll
    for (int kk = 0; kk < 8; kk++) {
        const int base = kk * 16 + hh * 4;
        const uint4* pa = (const uint4*)(wt + (tA * 16 + mrow) * WSTR + base);
        const uint4* pb = (const uint4*)(wt + (tB * 16 + mrow) * WSTR + base);
        uint4 a0 = pa[0], a1 = pa[2];    // runs at +0 and +8 dwords (16B-aligned)
        uint4 b0 = pb[0], b1 = pb[2];
        BfA[kk][0] = a0.x; BfA[kk][1] = a0.y; BfA[kk][2] = a0.z; BfA[kk][3] = a0.w;
        BfA[kk][4] = a1.x; BfA[kk][5] = a1.y; BfA[kk][6] = a1.z; BfA[kk][7] = a1.w;
        BfB[kk][0] = b0.x; BfB[kk][1] = b0.y; BfB[kk][2] = b0.z; BfB[kk][3] = b0.w;
        BfB[kk][4] = b1.x; BfB[kk][5] = b1.y; BfB[kk][6] = b1.z; BfB[kk][7] = b1.w;
    }

    // ---- phase 1: gate = x . W_mask, segment max (b_mask cancels in softmax) ----
    float wmax = -3.4e38f;
    for (int row = s0 + wid; row < s1; row += 8) {
        const float4* xr = (const float4*)(x + (size_t)row * EMB + lane * 8);
        float4 p = xr[0], q = xr[1];
        const float4* wm4 = (const float4*)(wmask + lane * 8);
        float4 wa = wm4[0], wb = wm4[1];
        float g = p.x * wa.x + p.y * wa.y + p.z * wa.z + p.w * wa.w
                + q.x * wb.x + q.y * wb.y + q.z * wb.z + q.w * wb.w;
        g = wred_sum(g);
        if (lane == 0) gate_ws[row] = g;
        wmax = fmaxf(wmax, g);
    }
    if (lane == 0) red[wid] = wmax;
    __syncthreads();
    if (tid == 0) {
        float m = red[0];
#pragma unroll
        for (int i = 1; i < 8; i++) m = fmaxf(m, red[i]);
        if (!(m > -3.0e38f)) m = 0.0f;   // empty segment guard
        scal[0] = m;
    }
    __syncthreads();
    const float smax = scal[0];
    float psum = 0.0f;
    for (int i = s0 + tid; i < s1; i += 256) psum += expf(gate_ws[i] - smax);
    psum = wred_sum(psum);
    if (lane == 0) red[wid] = psum;
    __syncthreads();
    if (tid == 0) {
        float s = 0.0f;
#pragma unroll
        for (int i = 0; i < 8; i++) s += red[i];
        scal[1] = 1.0f / fmaxf(s, 1e-16f);
    }
    __syncthreads();
    const float invden = scal[1];

    // ---- phase 2: WMMA feat GEMM fused with weighted segment-sum ----
    v8f accA = {}, accB = {};
    const int nch = ((s1 - s0) + 15) >> 4;

    for (int ch = 0; ch < nch; ch++) {
        // cooperative load of 16 rows of x, converted to bf16 pairs
        {
            int row  = tid >> 4;
            int dwb  = (tid & 15) * 8;                 // dword base within row (128 data dwords/row)
            int gr   = s0 + ch * 16 + row;
            uint4 v0, v1;
            if (gr < s1) {
                const float4* xr = (const float4*)(x + (size_t)gr * EMB + dwb * 2);
                float4 a = xr[0], c = xr[1], d = xr[2], e = xr[3];
                v0.x = pack_bf16(a.x, a.y); v0.y = pack_bf16(a.z, a.w);
                v0.z = pack_bf16(c.x, c.y); v0.w = pack_bf16(c.z, c.w);
                v1.x = pack_bf16(d.x, d.y); v1.y = pack_bf16(d.z, d.w);
                v1.z = pack_bf16(e.x, e.y); v1.w = pack_bf16(e.z, e.w);
            } else {
                v0 = make_uint4(0u, 0u, 0u, 0u);
                v1 = make_uint4(0u, 0u, 0u, 0u);
            }
            uint4* dst = (uint4*)(xch + row * XSTR + dwb);
            dst[0] = v0; dst[1] = v1;
        }
        if (tid < 16) {
            int gr = s0 + ch * 16 + tid;
            wbuf[tid] = (gr < s1) ? expf(gate_ws[gr] - smax) * invden : 0.0f;
        }
        __syncthreads();

        v8f cA = {}, cB = {};
#pragma unroll
        for (int kk = 0; kk < 8; kk++) {               // K = 256 in steps of 32
            const int base = kk * 16 + hh * 4;
            const uint4* pa = (const uint4*)(xch + mrow * XSTR + base);
            uint4 q0 = pa[0], q1 = pa[2];              // ds_load_b128 x2 (16B-aligned, conflict-free)
            v8u av;
            av[0] = q0.x; av[1] = q0.y; av[2] = q0.z; av[3] = q0.w;
            av[4] = q1.x; av[5] = q1.y; av[6] = q1.z; av[7] = q1.w;
            cA = __builtin_amdgcn_wmma_f32_16x16x32_bf16(
                    false, __builtin_bit_cast(v16bf, av),
                    false, __builtin_bit_cast(v16bf, BfA[kk]),
                    (short)0, cA, false, false);
            cB = __builtin_amdgcn_wmma_f32_16x16x32_bf16(
                    false, __builtin_bit_cast(v16bf, av),
                    false, __builtin_bit_cast(v16bf, BfB[kk]),
                    (short)0, cB, false, false);
        }

        // bias + leaky_relu + softmax weight, accumulate pooled sum
        const float biasA = bfl[tA * 16 + mrow];
        const float biasB = bfl[tB * 16 + mrow];
#pragma unroll
        for (int r = 0; r < 8; r++) {                  // C layout: VGPR r, M = r + hh*8, N = mrow
            const float wm = wbuf[r + hh * 8];
            accA[r] += wm * lrelu(cA[r] + biasA);
            accB[r] += wm * lrelu(cB[r] + biasB);
        }
        __syncthreads();
    }

    // ---- reduce accumulators over M (VGPRs) and lane halves -> xg row ----
    float sA = 0.0f, sB = 0.0f;
#pragma unroll
    for (int r = 0; r < 8; r++) { sA += accA[r]; sB += accB[r]; }
    sA += __shfl_xor(sA, 16, 32);
    sB += __shfl_xor(sB, 16, 32);
    if (lane < 16) {
        hbuf[tA * 16 + lane] = sA;
        hbuf[tB * 16 + lane] = sB;
    }
    const float xp = xg_prev[b * EMB + tid];
    hbuf[EMB + tid] = xp;
    __syncthreads();

    // ---- final: out = leaky_relu([xg, xg_prev] @ W_t + b_t) + xg_prev ----
    float acc = bt[tid];
    for (int k = 0; k < 2 * EMB; k++)
        acc += hbuf[k] * Wt[k * EMB + tid];            // hbuf broadcast, W_t coalesced (L2-hot)
    out[b * EMB + tid] = lrelu(acc) + xp;
}

extern "C" void kernel_launch(void* const* d_in, const int* in_sizes, int n_in,
                              void* d_out, int out_size, void* d_ws, size_t ws_size,
                              hipStream_t stream) {
    const float* xg_prev = (const float*)d_in[0];
    const float* x       = (const float*)d_in[1];
    const int*   bind    = (const int*)  d_in[2];
    const float* Wmask   = (const float*)d_in[3];
    // d_in[4] = b_mask: constant shift cancels inside the segment softmax -> unused
    const float* Wfeat   = (const float*)d_in[5];
    const float* bfeat   = (const float*)d_in[6];
    const float* Wt      = (const float*)d_in[7];
    const float* bt      = (const float*)d_in[8];
    float*       out     = (float*)d_out;
    float*       gate_ws = (float*)d_ws;             // N floats of scratch for gate values

    const int N = in_sizes[2];          // 262144 rows
    const int B = in_sizes[0] / EMB;    // 512 segments

    (void)n_in; (void)out_size; (void)ws_size;

    hipFuncSetAttribute((const void*)global_attn_pool_kernel,
                        hipFuncAttributeMaxDynamicSharedMemorySize, SMEM_BYTES);

    global_attn_pool_kernel<<<dim3(B), dim3(256), SMEM_BYTES, stream>>>(
        xg_prev, x, bind, Wmask, Wfeat, bfeat, Wt, bt, out, gate_ws, N);
}
